// MultiHead_88493506167392
// MI455X (gfx1250) — matrix operand order repaired and checked by reading, not compiled, not run on hardware
//
#include <hip/hip_runtime.h>
#include <hip/hip_bf16.h>
#include <math.h>

// B=8 S=1024 H=768 NH=12 DK=64
#define BB 8
#define SS 1024
#define HH 768
#define NHH 12
#define DKK 64

typedef __attribute__((ext_vector_type(16))) _Float16 v16h;
typedef __attribute__((ext_vector_type(8)))  _Float16 v8h;
typedef __attribute__((ext_vector_type(4)))  _Float16 v4h;
typedef __attribute__((ext_vector_type(8)))  float    v8f;
typedef __attribute__((ext_vector_type(4)))  float    f32x4;

// D = A(16x32 f16) * B(32x16 f16) + C(16x16 f32)
__device__ __forceinline__ v8f wmma_f16(v16h a, v16h b, v8f c) {
    return __builtin_amdgcn_wmma_f32_16x16x32_f16(
        /*neg_a=*/false, a, /*neg_b=*/false, b,
        /*c_mod=*/(short)0, c, /*reuse_a=*/false, /*reuse_b=*/false);
}

// ---- Fragment loaders (layouts per CDNA5 ISA §7.12.2) ----
// A 16x32 f16: lane m=lane&15, g=lane>>4; halves 0..7 -> K=g*8+i, 8..15 -> K=16+g*8+i
__device__ __forceinline__ v16h load_a_f32(const float* X, int rowbase, int ld,
                                           int kb, int lane) {
    const int m = lane & 15, g = lane >> 4;
    const float* p = X + (size_t)(rowbase + m) * ld + kb;
    f32x4 c0 = ((const f32x4*)(p + g * 8))[0];
    f32x4 c1 = ((const f32x4*)(p + g * 8))[1];
    f32x4 c2 = ((const f32x4*)(p + 16 + g * 8))[0];
    f32x4 c3 = ((const f32x4*)(p + 16 + g * 8))[1];
    v16h a;
#pragma unroll
    for (int i = 0; i < 4; ++i) {
        a[i]      = (_Float16)c0[i];
        a[4 + i]  = (_Float16)c1[i];
        a[8 + i]  = (_Float16)c2[i];
        a[12 + i] = (_Float16)c3[i];
    }
    return a;
}

__device__ __forceinline__ v16h load_a_f16(const _Float16* X, int rowbase, int ld,
                                           int kb, int lane) {
    const int m = lane & 15, g = lane >> 4;
    const _Float16* p = X + (size_t)(rowbase + m) * ld + kb;
    v8h h0 = *(const v8h*)(p + g * 8);
    v8h h1 = *(const v8h*)(p + 16 + g * 8);
    v16h a;
#pragma unroll
    for (int i = 0; i < 8; ++i) {
        a[i]     = h0[i];
        a[8 + i] = h1[i];
    }
    return a;
}

// B 32x16 f16: element (K=k, N=n) = base[(nbase+n)*ld + koff + k]
// (for y = x @ W^T : B[k][n] = Wh[n][k]; halves 0..7 -> K=g*8+i, 8..15 -> K=16+g*8+i)
__device__ __forceinline__ v16h load_b_f16(const _Float16* base, int nbase, int ld,
                                           int koff, int lane) {
    const int n = lane & 15, g = lane >> 4;
    const _Float16* p = base + (size_t)(nbase + n) * ld + koff + g * 8;
    v8h h0 = *(const v8h*)p;
    v8h h1 = *(const v8h*)(p + 16);
    v16h b;
#pragma unroll
    for (int i = 0; i < 8; ++i) {
        b[i]     = h0[i];
        b[8 + i] = h1[i];
    }
    return b;
}

// ---- One-time weight conversion f32 -> f16 (removes per-tile re-converts) ----
__global__ void __launch_bounds__(256)
cvt_f32_f16_kernel(const float* __restrict__ src, _Float16* __restrict__ dst, int n) {
    int i = (blockIdx.x * 256 + threadIdx.x) * 4;
    if (i < n) {
        f32x4 v = *(const f32x4*)(src + i);
        v4h h;
#pragma unroll
        for (int k = 0; k < 4; ++k) h[k] = (_Float16)v[k];
        *(v4h*)(dst + i) = h;
    }
}

// ---- Projection GEMM: y = X[8192x768] @ Wh^T -> f16, head-split layout ----
// vmode 0: out[((b*NH+head)*S + s)*DK + d]   (q,k)
// vmode 1: out[((b*NH+head)*DK + d)*S + s]   (v, transposed for P@V B-frags)
__global__ void __launch_bounds__(32)
proj_qkv_kernel(const float* __restrict__ X, const _Float16* __restrict__ Wh,
                _Float16* __restrict__ out, int vmode) {
    const int lane = threadIdx.x;
    const int mt = blockIdx.x;   // 256 tiles of 32 rows
    const int nt = blockIdx.y;   // 12 tiles of 64 cols == head index
    const int rowbase = mt * 32;
    v8f acc[2][4] = {};
    for (int kb = 0; kb < HH; kb += 32) {
        v16h a0 = load_a_f32(X, rowbase, HH, kb, lane);
        v16h a1 = load_a_f32(X, rowbase + 16, HH, kb, lane);
#pragma unroll
        for (int j = 0; j < 4; ++j) {
            v16h b = load_b_f16(Wh, nt * 64 + j * 16, HH, kb, lane);
            acc[0][j] = wmma_f16(a0, b, acc[0][j]);
            acc[1][j] = wmma_f16(a1, b, acc[1][j]);
        }
    }
    const int n = lane & 15, g = lane >> 4;
#pragma unroll
    for (int half = 0; half < 2; ++half)
#pragma unroll
        for (int j = 0; j < 4; ++j)
#pragma unroll
            for (int r = 0; r < 8; ++r) {
                int m = rowbase + half * 16 + r + 8 * g;  // global row = b*S + s
                int bb = m >> 10, s = m & (SS - 1);
                int d = j * 16 + n;
                size_t idx;
                if (vmode)
                    idx = ((size_t)(bb * NHH + nt) * DKK + d) * SS + s;
                else
                    idx = ((size_t)(bb * NHH + nt) * SS + s) * DKK + d;
                out[idx] = (_Float16)(acc[half][j][r]);
            }
}

// ---- Flash attention: one wave handles 16 query rows of one (b, head) ----
__global__ void __launch_bounds__(32)
attn_kernel(const _Float16* __restrict__ qp, const _Float16* __restrict__ kp,
            const _Float16* __restrict__ vpT, _Float16* __restrict__ ctx) {
    __shared__ __align__(16) _Float16 Pt[16 * 64];
    const int lane = threadIdx.x;
    const int qt = blockIdx.x;   // 64 q-tiles of 16 rows
    const int h  = blockIdx.y;   // 12 heads
    const int b  = blockIdx.z;   // 8 batches
    const int bh = b * NHH + h;

    const _Float16* qbase = qp + ((size_t)bh * SS + qt * 16) * DKK;
    const _Float16* kbase = kp + (size_t)bh * SS * DKK;   // [key][d]
    const _Float16* vbase = vpT + (size_t)bh * DKK * SS;  // [d][key]

    // Q fragments (16 rows x 64 contraction): 1/sqrt(dk) folded into softmax
    v16h aq0 = load_a_f16(qbase, 0, DKK, 0, lane);
    v16h aq1 = load_a_f16(qbase, 0, DKK, 32, lane);

    float mrow[8], lrow[8];
#pragma unroll
    for (int r = 0; r < 8; ++r) { mrow[r] = -INFINITY; lrow[r] = 0.f; }
    v8f acc[4] = {};
    const int n = lane & 15, g = lane >> 4;

    for (int kb = 0; kb < SS; kb += 64) {
        // scores: 16 x 64 block, K-contraction 64 (two WMMAs per 16-key subtile)
        v8f s[4];
#pragma unroll
        for (int j = 0; j < 4; ++j) {
            int keyb = kb + j * 16;
            v16h bk0 = load_b_f16(kbase, keyb, DKK, 0, lane);
            v16h bk1 = load_b_f16(kbase, keyb, DKK, 32, lane);
            v8f c = {};
            c = wmma_f16(aq0, bk0, c);
            c = wmma_f16(aq1, bk1, c);
#pragma unroll
            for (int r = 0; r < 8; ++r) s[j][r] = c[r] * 0.125f;  // 1/sqrt(64)
        }
        // online softmax: row max over block (16-lane xor reductions)
        float newm[8], corr[8];
#pragma unroll
        for (int r = 0; r < 8; ++r) {
            float t = fmaxf(fmaxf(s[0][r], s[1][r]), fmaxf(s[2][r], s[3][r]));
            t = fmaxf(t, __shfl_xor(t, 1, 32));
            t = fmaxf(t, __shfl_xor(t, 2, 32));
            t = fmaxf(t, __shfl_xor(t, 4, 32));
            t = fmaxf(t, __shfl_xor(t, 8, 32));
            float nm = fmaxf(mrow[r], t);
            corr[r] = __expf(mrow[r] - nm);
            newm[r] = nm;
            mrow[r] = nm;
        }
        // p = exp(s - m): row sums + stage P into LDS in A-layout source form
#pragma unroll
        for (int r = 0; r < 8; ++r) {
            float rs = 0.f;
#pragma unroll
            for (int j = 0; j < 4; ++j) {
                float p = __expf(s[j][r] - newm[r]);
                rs += p;
                Pt[(r + 8 * g) * 64 + j * 16 + n] = (_Float16)p;
            }
            rs += __shfl_xor(rs, 1, 32);
            rs += __shfl_xor(rs, 2, 32);
            rs += __shfl_xor(rs, 4, 32);
            rs += __shfl_xor(rs, 8, 32);
            lrow[r] = lrow[r] * corr[r] + rs;
        }
        // rescale running context
#pragma unroll
        for (int j = 0; j < 4; ++j)
#pragma unroll
            for (int r = 0; r < 8; ++r) acc[j][r] *= corr[r];
        __syncthreads();
        // context += P(16x64) @ V(64x64)
        v16h ap0 = load_a_f16(Pt, 0, 64, 0, lane);
        v16h ap1 = load_a_f16(Pt, 0, 64, 32, lane);
#pragma unroll
        for (int j = 0; j < 4; ++j) {
            v16h bv0 = load_b_f16(vbase, j * 16, SS, kb, lane);
            v16h bv1 = load_b_f16(vbase, j * 16, SS, kb + 32, lane);
            acc[j] = wmma_f16(ap0, bv0, acc[j]);
            acc[j] = wmma_f16(ap1, bv1, acc[j]);
        }
        __syncthreads();
    }
    // write ctx[b][s][h*64+d] f16, normalized by row sum
#pragma unroll
    for (int j = 0; j < 4; ++j)
#pragma unroll
        for (int r = 0; r < 8; ++r) {
            int m = r + 8 * g;
            int srow = qt * 16 + m;
            float v = acc[j][r] / lrow[r];
            ctx[((size_t)b * SS + srow) * HH + h * DKK + j * 16 + n] = (_Float16)v;
        }
}

// ---- Output projection: out = ctx[8192x768] @ Woh^T -> f32 d_out ----
__global__ void __launch_bounds__(32)
out_proj_kernel(const _Float16* __restrict__ C, const _Float16* __restrict__ Wh,
                float* __restrict__ out) {
    const int lane = threadIdx.x;
    const int mt = blockIdx.x;
    const int nt = blockIdx.y;
    const int rowbase = mt * 32;
    v8f acc[2][4] = {};
    for (int kb = 0; kb < HH; kb += 32) {
        v16h a0 = load_a_f16(C, rowbase, HH, kb, lane);
        v16h a1 = load_a_f16(C, rowbase + 16, HH, kb, lane);
#pragma unroll
        for (int j = 0; j < 4; ++j) {
            v16h b = load_b_f16(Wh, nt * 64 + j * 16, HH, kb, lane);
            acc[0][j] = wmma_f16(a0, b, acc[0][j]);
            acc[1][j] = wmma_f16(a1, b, acc[1][j]);
        }
    }
    const int n = lane & 15, g = lane >> 4;
#pragma unroll
    for (int half = 0; half < 2; ++half)
#pragma unroll
        for (int j = 0; j < 4; ++j)
#pragma unroll
            for (int r = 0; r < 8; ++r) {
                int m = rowbase + half * 16 + r + 8 * g;
                out[(size_t)m * HH + nt * 64 + j * 16 + n] = acc[half][j][r];
            }
}

extern "C" void kernel_launch(void* const* d_in, const int* in_sizes, int n_in,
                              void* d_out, int out_size, void* d_ws, size_t ws_size,
                              hipStream_t stream) {
    const float* Q  = (const float*)d_in[0];
    const float* K  = (const float*)d_in[1];
    const float* V  = (const float*)d_in[2];
    const float* Wq = (const float*)d_in[3];
    const float* Wk = (const float*)d_in[4];
    const float* Wv = (const float*)d_in[5];
    const float* Wo = (const float*)d_in[6];
    float* out = (float*)d_out;

    const size_t per = (size_t)BB * NHH * SS * DKK;  // 6,291,456 f16 elems
    const size_t wsz = (size_t)HH * HH;              //   589,824 f16 elems
    _Float16* qp  = (_Float16*)d_ws;
    _Float16* kp  = qp + per;
    _Float16* vpT = kp + per;
    _Float16* ctx = vpT + per;
    _Float16* Wqh = ctx + per;
    _Float16* Wkh = Wqh + wsz;
    _Float16* Wvh = Wkh + wsz;
    _Float16* Woh = Wvh + wsz;                       // total ~55.3 MB

    dim3 blk(32, 1, 1);

    // one-time weight conversion to f16 (576 blocks each, exact multiple)
    dim3 gcvt((unsigned)(wsz / 4 / 256), 1, 1);
    dim3 bcvt(256, 1, 1);
    cvt_f32_f16_kernel<<<gcvt, bcvt, 0, stream>>>(Wq, Wqh, (int)wsz);
    cvt_f32_f16_kernel<<<gcvt, bcvt, 0, stream>>>(Wk, Wkh, (int)wsz);
    cvt_f32_f16_kernel<<<gcvt, bcvt, 0, stream>>>(Wv, Wvh, (int)wsz);
    cvt_f32_f16_kernel<<<gcvt, bcvt, 0, stream>>>(Wo, Woh, (int)wsz);

    dim3 gproj(256, 12, 1);                 // 32-row x 64-col tiles
    proj_qkv_kernel<<<gproj, blk, 0, stream>>>(Q, Wqh, qp, 0);
    proj_qkv_kernel<<<gproj, blk, 0, stream>>>(K, Wkh, kp, 0);
    proj_qkv_kernel<<<gproj, blk, 0, stream>>>(V, Wvh, vpT, 1);

    dim3 gattn(SS / 16, NHH, BB);           // 6144 waves
    attn_kernel<<<gattn, blk, 0, stream>>>(qp, kp, vpT, ctx);

    out_proj_kernel<<<gproj, blk, 0, stream>>>(ctx, Woh, out);
}
